// WanSelfAttention_88399016886748
// MI455X (gfx1250) — compile-verified, matrix operand order checked by reading, not run on hardware
//
#include <hip/hip_runtime.h>

// ---------------------------------------------------------------- constants
#define SEQ   2048
#define DIMN  1536
#define HEADS 12
#define HDIM  128
#define FCOLS 64          // freqs is (1024, 64)
#define TOPKK 204         // int(0.1 * 2048)

typedef __bf16 bf16_t;
typedef __attribute__((ext_vector_type(16))) __bf16 v16bf;
typedef __attribute__((ext_vector_type(8)))  __bf16 v8bf;
typedef __attribute__((ext_vector_type(8)))  float  v8f;
typedef __attribute__((ext_vector_type(4)))  unsigned int u32x4;
typedef __attribute__((ext_vector_type(8)))  int i32x8;
typedef __attribute__((ext_vector_type(4)))  int i32x4;
typedef int gv4i __attribute__((vector_size(16)));   // matches builtin pointer params

// ---------------------------------------------------------------- CDNA5 sync macros
#if __has_builtin(__builtin_amdgcn_s_wait_asynccnt)
  #define WAIT_ASYNCCNT(n) __builtin_amdgcn_s_wait_asynccnt(n)
#else
  #define WAIT_ASYNCCNT(n) asm volatile("s_wait_asynccnt %0" ::"n"(n) : "memory")
#endif
#if __has_builtin(__builtin_amdgcn_s_wait_tensorcnt)
  #define WAIT_TENSORCNT(n) __builtin_amdgcn_s_wait_tensorcnt(n)
#else
  #define WAIT_TENSORCNT(n) asm volatile("s_wait_tensorcnt %0" ::"n"(n) : "memory")
#endif
#if __has_builtin(__builtin_amdgcn_s_wait_dscnt)
  #define WAIT_DSCNT(n) __builtin_amdgcn_s_wait_dscnt(n)
#else
  #define WAIT_DSCNT(n) asm volatile("s_wait_dscnt %0" ::"n"(n) : "memory")
#endif

// ---------------------------------------------------------------- async global->LDS copy
#if __has_builtin(__builtin_amdgcn_global_load_async_to_lds_b128)
#define HAVE_ASYNC_LDS 1
__device__ __forceinline__ void async_copy_b128(const bf16_t* gsrc, bf16_t* ldst) {
  __builtin_amdgcn_global_load_async_to_lds_b128(
      (__attribute__((address_space(1))) gv4i*)gsrc,
      (__attribute__((address_space(3))) gv4i*)ldst, 0, 0);
}
#else
#define HAVE_ASYNC_LDS 0
__device__ __forceinline__ void async_copy_b128(const bf16_t* gsrc, bf16_t* ldst) {
  *reinterpret_cast<v8bf*>(ldst) = *reinterpret_cast<const v8bf*>(gsrc);
}
#endif

// ---------------------------------------------------------------- helpers
__device__ __forceinline__ v16bf load_frag16(const bf16_t* p) {
  // per-lane A/B fragment: 8 bf16 at p (K chunk 0) and 8 bf16 at p+16 (K chunk +16)
  v8bf lo = *reinterpret_cast<const v8bf*>(p);
  v8bf hi = *reinterpret_cast<const v8bf*>(p + 16);
  v16bf r;
#pragma unroll
  for (int i = 0; i < 8; ++i) { r[i] = lo[i]; r[i + 8] = hi[i]; }
  return r;
}

__device__ __forceinline__ float wave_red_sum(float v) {
#pragma unroll
  for (int m = 16; m > 0; m >>= 1) v += __shfl_xor(v, m, 32);
  return v;
}
__device__ __forceinline__ float wave_red_max(float v) {
#pragma unroll
  for (int m = 16; m > 0; m >>= 1) v = fmaxf(v, __shfl_xor(v, m, 32));
  return v;
}
__device__ __forceinline__ int wave_red_sumi(int v) {
#pragma unroll
  for (int m = 16; m > 0; m >>= 1) v += __shfl_xor(v, m, 32);
  return v;
}

// ---------------------------------------------------------------- fp32 -> bf16
__global__ __launch_bounds__(256) void cvt_bf16_kernel(const float* __restrict__ in,
                                                       bf16_t* __restrict__ out, int n) {
  int i = blockIdx.x * 256 + threadIdx.x;
  if (i < n) out[i] = (bf16_t)in[i];
}

// ---------------------------------------------------------------- GEMM: C = A @ W^T + b
// A: (SEQ x DIMN) bf16 row-major.  W: (DIMN x DIMN) bf16 row-major (used as W^T).
// grid = (SEQ/16, DIMN/128), block = 256 (8 waves, wave w -> N offset w*16).
__global__ __launch_bounds__(256) void gemm_bf16_kernel(const bf16_t* __restrict__ A,
                                                        const bf16_t* __restrict__ W,
                                                        const float* __restrict__ bias,
                                                        float* __restrict__ C) {
  const int lane = threadIdx.x & 31;
  const int wave = threadIdx.x >> 5;
  const int l16  = lane & 15;
  const int half = lane >> 4;
  const int m0 = blockIdx.x * 16;
  const int n0 = blockIdx.y * 128 + wave * 16;

  v8f acc;
  const float bv = bias[n0 + l16];
#pragma unroll
  for (int r = 0; r < 8; ++r) acc[r] = bv;

  const bf16_t* arow = A + (size_t)(m0 + l16) * DIMN + half * 8;
  const bf16_t* brow = W + (size_t)(n0 + l16) * DIMN + half * 8;

#pragma unroll 4
  for (int kb = 0; kb < DIMN; kb += 32) {
    __builtin_prefetch(arow + kb + 128, 0, 1);   // -> global_prefetch_b8
    __builtin_prefetch(brow + kb + 128, 0, 1);
    v16bf af = load_frag16(arow + kb);
    v16bf bf = load_frag16(brow + kb);
    acc = __builtin_amdgcn_wmma_f32_16x16x32_bf16(false, af, false, bf,
                                                  (short)0, acc, false, false);
  }
#pragma unroll
  for (int r = 0; r < 8; ++r)
    C[(size_t)(m0 + r + 8 * half) * DIMN + n0 + l16] = acc[r];
}

// ---------------------------------------------------------------- RMSNorm + RoPE + V transpose
__global__ __launch_bounds__(256) void norm_rope_kernel(const float* __restrict__ Qf,
                                                        const float* __restrict__ Kf,
                                                        const float* __restrict__ Vf,
                                                        const float* __restrict__ gq,
                                                        const float* __restrict__ gk,
                                                        const float* __restrict__ freqs,
                                                        bf16_t* __restrict__ Qb,
                                                        bf16_t* __restrict__ Kb,
                                                        bf16_t* __restrict__ Vt) {
  __shared__ float redbuf[16];
  __shared__ float bcast[2];
  const int s = blockIdx.x;
  const int t = threadIdx.x;
  const int lane = t & 31, wave = t >> 5;

  const float* qrow = Qf + (size_t)s * DIMN;
  const float* krow = Kf + (size_t)s * DIMN;

  float sq = 0.f, sk = 0.f;
  for (int i = t; i < DIMN; i += 256) {
    float a = qrow[i]; sq += a * a;
    float b = krow[i]; sk += b * b;
  }
  sq = wave_red_sum(sq);
  sk = wave_red_sum(sk);
  if (lane == 0) { redbuf[wave] = sq; redbuf[8 + wave] = sk; }
  __syncthreads();
  if (t == 0) {
    float a = 0.f, b = 0.f;
    for (int i = 0; i < 8; ++i) { a += redbuf[i]; b += redbuf[8 + i]; }
    bcast[0] = rsqrtf(a / (float)DIMN + 1e-6f);
    bcast[1] = rsqrtf(b / (float)DIMN + 1e-6f);
  }
  __syncthreads();
  const float rq = bcast[0], rk = bcast[1];

  const int fi = s >> 10, hi = (s >> 5) & 31, wi = s & 31;
  for (int p = t; p < DIMN / 2; p += 256) {
    const int j = p & 63;                              // pair index within head
    const int pos = (j < 22) ? fi : ((j < 43) ? hi : wi);
    const float ang = freqs[pos * FCOLS + j];
    float sn, cs;
    __sincosf(ang, &sn, &cs);
    const int ei = 2 * p, oi = 2 * p + 1;
    const float qe = qrow[ei] * rq * gq[ei], qo = qrow[oi] * rq * gq[oi];
    Qb[(size_t)s * DIMN + ei] = (bf16_t)(qe * cs - qo * sn);
    Qb[(size_t)s * DIMN + oi] = (bf16_t)(qe * sn + qo * cs);
    const float ke = krow[ei] * rk * gk[ei], ko = krow[oi] * rk * gk[oi];
    Kb[(size_t)s * DIMN + ei] = (bf16_t)(ke * cs - ko * sn);
    Kb[(size_t)s * DIMN + oi] = (bf16_t)(ke * sn + ko * cs);
  }
  const float* vrow = Vf + (size_t)s * DIMN;
  for (int i = t; i < DIMN; i += 256)
    Vt[(size_t)i * SEQ + s] = (bf16_t)vrow[i];         // d-major V for PV B-fragments
}

// ---------------------------------------------------------------- attention
// grid = (SEQ/16, HEADS), block = 256.  Dynamic LDS (CDNA5: 320 KB/WGP):
//   sc:  16 x 2048 fp32 score panel       (131072 B)
//   wt:  16 x 2048 bf16 weight panel      ( 65536 B)
//   qt:  16 x 128  bf16 q tile            (  4096 B)  <- TDM tensor_load_to_lds
//   kt:  8 waves x 2 bufs x 16x128 bf16   ( 65536 B)  <- async global->LDS double buffer
#define SC_OFF 0
#define WT_OFF (16 * SEQ * 4)
#define QT_OFF (16 * SEQ * 6)
#define KT_OFF (16 * SEQ * 6 + 16 * HDIM * 2)
#define ATTN_SMEM (KT_OFF + 8 * 2 * 16 * HDIM * 2)   // 266240 bytes

__global__ __launch_bounds__(256) void attn_topk_kernel(const bf16_t* __restrict__ Qb,
                                                        const bf16_t* __restrict__ Kb,
                                                        const bf16_t* __restrict__ Vt,
                                                        bf16_t* __restrict__ Ob) {
  extern __shared__ char smem_raw[];
  float*  sc = reinterpret_cast<float*>(smem_raw + SC_OFF);
  bf16_t* wt = reinterpret_cast<bf16_t*>(smem_raw + WT_OFF);
  bf16_t* qt = reinterpret_cast<bf16_t*>(smem_raw + QT_OFF);
  bf16_t* kt = reinterpret_cast<bf16_t*>(smem_raw + KT_OFF);

  const int qb   = blockIdx.x;          // q tile (16 rows)
  const int head = blockIdx.y;
  const int t    = threadIdx.x;
  const int lane = t & 31;
  const int wave = t >> 5;
  const int l16  = lane & 15;
  const int half = lane >> 4;

  // -------- stage q tile (16 x 128 bf16) via Tensor Data Mover
#if __has_builtin(__builtin_amdgcn_tensor_load_to_lds)
  if (wave == 0) {
    const unsigned long long ga =
        (unsigned long long)(Qb + (size_t)(qb * 16) * DIMN + head * HDIM);
    // No static LDS in this kernel -> dynamic LDS base is offset 0.
    const unsigned lds_addr = QT_OFF;
    u32x4 g0;
    g0[0] = 1u;                                    // count=1, gather off
    g0[1] = lds_addr;                              // lds_addr
    g0[2] = (unsigned)(ga & 0xFFFFFFFFu);          // global_addr[31:0]
    g0[3] = (unsigned)((ga >> 32) & 0x01FFFFFFu) | (2u << 30);  // addr[56:32] | type=2
    i32x8 g1;
    g1[0] = (1 << 16);                             // data_size=2B, no mask/flags
    g1[1] = (int)((DIMN & 0xFFFF) << 16);          // tensor_dim0 lo16 @ bits63:48
    g1[2] = (int)((DIMN >> 16) | ((SEQ & 0xFFFF) << 16));  // dim0 hi, tensor_dim1 lo
    g1[3] = (int)((SEQ >> 16) | (HDIM << 16));     // dim1 hi, tile_dim0=128
    g1[4] = 16;                                    // tile_dim1=16, tile_dim2=0
    g1[5] = DIMN;                                  // tensor_dim0_stride lo32
    g1[6] = 0;
    g1[7] = 0;
    i32x4 z4 = {};
    i32x8 z8 = {};
    __builtin_amdgcn_tensor_load_to_lds(g0, g1, z4, z4, z8, 0);  // 6-arg flavor
    WAIT_TENSORCNT(0);
  }
#else
  for (int idx = t; idx < 16 * HDIM; idx += 256) {
    const int r = idx >> 7, c = idx & 127;
    qt[idx] = Qb[(size_t)(qb * 16 + r) * DIMN + head * HDIM + c];
  }
#endif
  __syncthreads();

  // -------- scores: each wave owns key-blocks kb = wave + 8*it, K tiles staged
  //          into a per-wave LDS double buffer via async global->LDS loads.
  bf16_t* ktw = kt + wave * (2 * 16 * HDIM);      // this wave's 2 buffers
  auto issue_ktile = [&](int kb, int buf) {
    const bf16_t* src =
        Kb + (size_t)(kb * 16 + (lane >> 1)) * DIMN + head * HDIM + (lane & 1) * 64;
    bf16_t* dst = ktw + buf * (16 * HDIM) + (lane >> 1) * HDIM + (lane & 1) * 64;
#pragma unroll
    for (int i = 0; i < 8; ++i) async_copy_b128(src + i * 8, dst + i * 8);
  };

  const float inv_sqrt_d = 0.088388347648318447f;   // 1/sqrt(128)
  issue_ktile(wave, 0);
  for (int it = 0; it < 16; ++it) {
    const int kb = wave + it * 8;
    const int buf = it & 1;
    if (it + 1 < 16) {
      WAIT_DSCNT(0);                 // don't overwrite a buffer still being read
      issue_ktile(kb + 8, buf ^ 1);  // prefetch next K tile
      WAIT_ASYNCCNT(8);              // current buffer's 8 async loads done
    } else {
      WAIT_ASYNCCNT(0);
    }
    v8f acc = {};
    const bf16_t* qrow = qt + l16 * HDIM + half * 8;
    const bf16_t* krow = ktw + buf * (16 * HDIM) + l16 * HDIM + half * 8;
#pragma unroll
    for (int dc = 0; dc < 4; ++dc) {
      v16bf af = load_frag16(qrow + dc * 32);
      v16bf bf = load_frag16(krow + dc * 32);
      acc = __builtin_amdgcn_wmma_f32_16x16x32_bf16(false, af, false, bf,
                                                    (short)0, acc, false, false);
    }
#pragma unroll
    for (int r = 0; r < 8; ++r)
      sc[(r + 8 * half) * SEQ + kb * 16 + l16] = acc[r] * inv_sqrt_d;
  }
  __syncthreads();

  // -------- per-row top-k (~204) threshold + softmax -> bf16 weights
  for (int rr = 0; rr < 2; ++rr) {
    const int r = wave + rr * 8;
    const float* row = sc + (size_t)r * SEQ;
    float vmax = -1e30f, vmin = 1e30f;
    for (int i = 0; i < SEQ / 32; ++i) {
      const float v = row[lane + i * 32];
      vmax = fmaxf(vmax, v);
      vmin = fminf(vmin, v);
    }
    vmax = wave_red_max(vmax);
    vmin = -wave_red_max(-vmin);

    float lo = vmin, hi = vmax;
    for (int it = 0; it < 20; ++it) {           // binary search the k-th value
      const float mid = 0.5f * (lo + hi);
      int cnt = 0;
      for (int i = 0; i < SEQ / 32; ++i)
        cnt += (row[lane + i * 32] >= mid) ? 1 : 0;
      cnt = wave_red_sumi(cnt);
      if (cnt > TOPKK) lo = mid; else hi = mid;
    }
    const float thresh = lo;

    float ssum = 0.f;
    for (int i = 0; i < SEQ / 32; ++i) {
      const float v = row[lane + i * 32];
      if (v >= thresh) ssum += __expf(v - vmax);
    }
    ssum = wave_red_sum(ssum);
    const float inv = 1.0f / ssum;
    for (int i = 0; i < SEQ / 32; ++i) {
      const int idx = lane + i * 32;
      const float v = row[idx];
      wt[(size_t)r * SEQ + idx] =
          (v >= thresh) ? (bf16_t)(__expf(v - vmax) * inv) : (bf16_t)0.0f;
    }
  }
  __syncthreads();

  // -------- out tile = weights(16x2048) @ V(2048x128); wave w owns d-cols w*16..
  {
    v8f acc = {};
    const bf16_t* wrow = wt + (size_t)l16 * SEQ + half * 8;
    const bf16_t* vcol = Vt + (size_t)(head * HDIM + wave * 16 + l16) * SEQ + half * 8;
#pragma unroll 4
    for (int kb = 0; kb < SEQ; kb += 32) {
      v16bf af = load_frag16(wrow + kb);
      v16bf bf = load_frag16(vcol + kb);
      acc = __builtin_amdgcn_wmma_f32_16x16x32_bf16(false, af, false, bf,
                                                    (short)0, acc, false, false);
    }
#pragma unroll
    for (int r = 0; r < 8; ++r)
      Ob[(size_t)(qb * 16 + r + 8 * half) * DIMN + head * HDIM + wave * 16 + l16] =
          (bf16_t)acc[r];
  }
}

// ---------------------------------------------------------------- host launcher
extern "C" void kernel_launch(void* const* d_in, const int* in_sizes, int n_in,
                              void* d_out, int out_size, void* d_ws, size_t ws_size,
                              hipStream_t stream) {
  const float* x     = (const float*)d_in[0];
  // d_in[1]=seq_lens, d_in[2]=grid_sizes (static here: 2048, (2,32,32))
  const float* freqs = (const float*)d_in[3];
  const float* Wq = (const float*)d_in[4];
  const float* bq = (const float*)d_in[5];
  const float* Wk = (const float*)d_in[6];
  const float* bk = (const float*)d_in[7];
  const float* Wv = (const float*)d_in[8];
  const float* bv = (const float*)d_in[9];
  const float* Wo = (const float*)d_in[10];
  const float* bo = (const float*)d_in[11];
  const float* gq = (const float*)d_in[12];
  const float* gk = (const float*)d_in[13];
  float* out = (float*)d_out;

  char* ws = (char*)d_ws;
  size_t off = 0;
  auto alloc = [&](size_t bytes) -> char* {
    char* p = ws + off;
    off += (bytes + 255) & ~(size_t)255;
    return p;
  };
  const size_t SD = (size_t)SEQ * DIMN;
  const size_t DD = (size_t)DIMN * DIMN;

  bf16_t* xb  = (bf16_t*)alloc(SD * 2);
  bf16_t* wqb = (bf16_t*)alloc(DD * 2);
  bf16_t* wkb = (bf16_t*)alloc(DD * 2);
  bf16_t* wvb = (bf16_t*)alloc(DD * 2);
  bf16_t* wob = (bf16_t*)alloc(DD * 2);
  float*  Qf  = (float*)alloc(SD * 4);
  float*  Kf  = (float*)alloc(SD * 4);
  float*  Vf  = (float*)alloc(SD * 4);
  bf16_t* Qb  = (bf16_t*)alloc(SD * 2);
  bf16_t* Kb  = (bf16_t*)alloc(SD * 2);
  bf16_t* Vt  = (bf16_t*)alloc(SD * 2);
  bf16_t* Ob  = (bf16_t*)alloc(SD * 2);

  // 1) fp32 -> bf16 conversions
  cvt_bf16_kernel<<<(int)((SD + 255) / 256), 256, 0, stream>>>(x, xb, (int)SD);
  cvt_bf16_kernel<<<(int)((DD + 255) / 256), 256, 0, stream>>>(Wq, wqb, (int)DD);
  cvt_bf16_kernel<<<(int)((DD + 255) / 256), 256, 0, stream>>>(Wk, wkb, (int)DD);
  cvt_bf16_kernel<<<(int)((DD + 255) / 256), 256, 0, stream>>>(Wv, wvb, (int)DD);
  cvt_bf16_kernel<<<(int)((DD + 255) / 256), 256, 0, stream>>>(Wo, wob, (int)DD);

  // 2) Q/K/V projections (WMMA GEMM, fp32 out)
  dim3 gg(SEQ / 16, DIMN / 128);
  gemm_bf16_kernel<<<gg, 256, 0, stream>>>(xb, wqb, bq, Qf);
  gemm_bf16_kernel<<<gg, 256, 0, stream>>>(xb, wkb, bk, Kf);
  gemm_bf16_kernel<<<gg, 256, 0, stream>>>(xb, wvb, bv, Vf);

  // 3) RMSNorm + RoPE + V transpose -> bf16 operands
  norm_rope_kernel<<<SEQ, 256, 0, stream>>>(Qf, Kf, Vf, gq, gk, freqs, Qb, Kb, Vt);

  // 4) sparse top-k attention (TDM q-tile + async K double-buffer; 260 KB LDS)
  attn_topk_kernel<<<dim3(SEQ / 16, HEADS), 256, ATTN_SMEM, stream>>>(Qb, Kb, Vt, Ob);

  // 5) output projection -> d_out
  gemm_bf16_kernel<<<gg, 256, 0, stream>>>(Ob, wob, bo, out);
}
